// SparseGraphAttention_46325517254953
// MI455X (gfx1250) — matrix operand order repaired and checked by reading, not compiled, not run on hardware
//
#include <hip/hip_runtime.h>

// ---------------------------------------------------------------------------
// Sparse Graph Attention (GAT) for MI455X / gfx1250, wave32.
// h = x@W via v_wmma_f32_16x16x32_f16; rest is gather/scatter + global softmax.
// ---------------------------------------------------------------------------

typedef __attribute__((ext_vector_type(16))) _Float16 v16h;
typedef __attribute__((ext_vector_type(8)))  _Float16 v8h;
typedef __attribute__((ext_vector_type(8)))  float    v8f;

#define N_NODES 50000
#define N_EDGES 800000
#define IN_DIM  128
#define OUT_DIM 64
#define ALPHA   0.2f

// order-preserving float <-> uint mapping (for atomicMax on floats incl. negatives)
__device__ __forceinline__ unsigned f32_to_ord(float f) {
  unsigned u = __float_as_uint(f);
  return (u & 0x80000000u) ? ~u : (u | 0x80000000u);
}
__device__ __forceinline__ float ord_to_f32(unsigned u) {
  unsigned b = (u & 0x80000000u) ? (u ^ 0x80000000u) : ~u;
  return __uint_as_float(b);
}

// ---------------------------------------------------------------------------
// 0) init: zero h_prime accumulator, seed global max / sum scalars
// ---------------------------------------------------------------------------
__global__ __launch_bounds__(256) void gat_init_kernel(float* __restrict__ hp,
                                                       unsigned* __restrict__ maxbits,
                                                       float* __restrict__ sum) {
  int i = blockIdx.x * 256 + threadIdx.x;            // grid covers N*OUT exactly
  hp[i] = 0.0f;
  if (i == 0) {
    *maxbits = f32_to_ord(-3.4e38f);
    *sum = 0.0f;
  }
}

// ---------------------------------------------------------------------------
// 1) GEMM: h[N,64] = x[N,128] @ W[128,64], f16 WMMA, f32 accumulate.
//    Block = 128 threads = 4 waves; block handles 16 rows; wave w handles
//    column tile [16w, 16w+16). K=128 -> 4 x wmma_f32_16x16x32_f16.
// ---------------------------------------------------------------------------
__global__ __launch_bounds__(128) void gat_gemm_kernel(const float* __restrict__ x,
                                                       const float* __restrict__ W,
                                                       float* __restrict__ h) {
  __shared__ alignas(16) _Float16 xs[16][IN_DIM];       // 4 KB: x tile as f16
  __shared__ alignas(16) _Float16 Wt[OUT_DIM][IN_DIM];  // 16 KB: W^T as f16

  const int t    = threadIdx.x;
  const int row0 = blockIdx.x * 16;

  // stage x tile (16x128), coalesced: consecutive threads -> consecutive cols
#pragma unroll
  for (int i = 0; i < 16; ++i) {
    int idx = t + i * 128;
    int r = idx >> 7, c = idx & 127;
    xs[r][c] = (_Float16)x[(row0 + r) * IN_DIM + c];
  }
  // stage W transposed (Wt[col][k]); global read coalesced over cols
#pragma unroll
  for (int i = 0; i < 64; ++i) {
    int idx = t + i * 128;
    int k = idx >> 6, c = idx & 63;
    Wt[c][k] = (_Float16)W[k * OUT_DIM + c];
  }
  __syncthreads();

  const int lane = t & 31;
  const int wave = t >> 5;        // 0..3 -> column tile
  const int l    = lane & 15;
  const int hi   = lane >> 4;     // which half-wave
  const int n0   = wave * 16;

  v8f acc = {};
#pragma unroll
  for (int kc = 0; kc < 4; ++kc) {
    // A fragment (16x32 f16): lane row M=l; halves 0..7 = K klo..klo+7,
    // halves 8..15 = K klo+16..klo+23, klo = kc*32 + hi*8  (ISA 7.12.2)
    const int klo = kc * 32 + hi * 8;
    v8h a0 = *(const v8h*)&xs[l][klo];
    v8h a1 = *(const v8h*)&xs[l][klo + 16];
    // B fragment (32x16 f16): lane col N=l; halves 0..7 = K kb..kb+7,
    // halves 8..15 = K kb+16..kb+23, kb = kc*32 + hi*8
    const int kb = kc * 32 + hi * 8;
    v8h b0 = *(const v8h*)&Wt[n0 + l][kb];
    v8h b1 = *(const v8h*)&Wt[n0 + l][kb + 16];
    v16h A, B;
#pragma unroll
    for (int i = 0; i < 8; ++i) {
      A[i] = a0[i]; A[8 + i] = a1[i];
      B[i] = b0[i]; B[8 + i] = b1[i];
    }
    acc = __builtin_amdgcn_wmma_f32_16x16x32_f16(
        /*neg_a=*/false, A, /*neg_b=*/false, B,
        /*c_mod=*/(short)0, acc, /*reuse_a=*/false, /*reuse_b=*/false);
  }

  // C/D layout: VGPR v holds row M = v + 8*hi, column N = l (ISA 7.12.2)
#pragma unroll
  for (int v = 0; v < 8; ++v) {
    int m = v + hi * 8;
    h[(row0 + m) * OUT_DIM + n0 + l] = acc[v];
  }
}

// ---------------------------------------------------------------------------
// 2) node scores: a_src[n] = h[n,:]@a[0:64], a_dst[n] = h[n,:]@a[64:128]
//    one wave per node, wave32 shuffle reduction
// ---------------------------------------------------------------------------
__global__ __launch_bounds__(256) void gat_node_score_kernel(const float* __restrict__ h,
                                                             const float* __restrict__ a,
                                                             float* __restrict__ asrc,
                                                             float* __restrict__ adst) {
  int node = (blockIdx.x * 256 + threadIdx.x) >> 5;  // grid covers N exactly
  int lane = threadIdx.x & 31;
  float h0 = h[node * OUT_DIM + lane];
  float h1 = h[node * OUT_DIM + lane + 32];
  float ps = h0 * a[lane]      + h1 * a[lane + 32];
  float pd = h0 * a[64 + lane] + h1 * a[96 + lane];
#pragma unroll
  for (int m = 16; m > 0; m >>= 1) {
    ps += __shfl_xor(ps, m, 32);
    pd += __shfl_xor(pd, m, 32);
  }
  if (lane == 0) { asrc[node] = ps; adst[node] = pd; }
}

// ---------------------------------------------------------------------------
// 3) edge scores + global max (pass 1 of softmax)
// ---------------------------------------------------------------------------
__global__ __launch_bounds__(256) void gat_edge_max_kernel(const int* __restrict__ src,
                                                           const int* __restrict__ dst,
                                                           const float* __restrict__ asrc,
                                                           const float* __restrict__ adst,
                                                           float* __restrict__ e,
                                                           unsigned* __restrict__ maxbits) {
  __shared__ float sred[8];
  int i = blockIdx.x * 256 + threadIdx.x;            // grid covers E exactly
  float s = asrc[src[i]] + adst[dst[i]];
  float v = s > 0.0f ? s : ALPHA * s;                // leaky relu
  e[i] = v;
  // warp max
#pragma unroll
  for (int m = 16; m > 0; m >>= 1) v = fmaxf(v, __shfl_xor(v, m, 32));
  if ((threadIdx.x & 31) == 0) sred[threadIdx.x >> 5] = v;
  __syncthreads();
  if (threadIdx.x < 8) {
    float w = sred[threadIdx.x];
#pragma unroll
    for (int m = 4; m > 0; m >>= 1) w = fmaxf(w, __shfl_xor(w, m, 32));
    if (threadIdx.x == 0) atomicMax(maxbits, f32_to_ord(w));
  }
}

// ---------------------------------------------------------------------------
// 4) exp(e - max) in place + global sum (pass 2 of softmax)
// ---------------------------------------------------------------------------
__global__ __launch_bounds__(256) void gat_edge_exp_kernel(float* __restrict__ e,
                                                           const unsigned* __restrict__ maxbits,
                                                           float* __restrict__ sum) {
  __shared__ float sred[8];
  int i = blockIdx.x * 256 + threadIdx.x;
  float mx = ord_to_f32(*maxbits);
  float w = __expf(e[i] - mx);
  e[i] = w;
#pragma unroll
  for (int m = 16; m > 0; m >>= 1) w += __shfl_xor(w, m, 32);
  if ((threadIdx.x & 31) == 0) sred[threadIdx.x >> 5] = w;
  __syncthreads();
  if (threadIdx.x < 8) {
    float s = sred[threadIdx.x];
#pragma unroll
    for (int m = 4; m > 0; m >>= 1) s += __shfl_xor(s, m, 32);
    if (threadIdx.x == 0) atomicAdd(sum, s);
  }
}

// ---------------------------------------------------------------------------
// 5) scatter: hp[src[e], :] += (w[e]/sum) * h[dst[e], :]; one wave per edge
// ---------------------------------------------------------------------------
__global__ __launch_bounds__(256) void gat_scatter_kernel(const int* __restrict__ src,
                                                          const int* __restrict__ dst,
                                                          const float* __restrict__ w,
                                                          const float* __restrict__ sum,
                                                          const float* __restrict__ h,
                                                          float* __restrict__ hp) {
  int edge = blockIdx.x * 8 + (threadIdx.x >> 5);    // grid covers E exactly
  int lane = threadIdx.x & 31;
  float att = w[edge] * (1.0f / *sum);
  int s = src[edge];
  int d = dst[edge];
  atomicAdd(&hp[s * OUT_DIM + lane],      att * h[d * OUT_DIM + lane]);
  atomicAdd(&hp[s * OUT_DIM + lane + 32], att * h[d * OUT_DIM + lane + 32]);
}

// ---------------------------------------------------------------------------
// 6) elu
// ---------------------------------------------------------------------------
__global__ __launch_bounds__(256) void gat_elu_kernel(const float* __restrict__ hp,
                                                      float* __restrict__ out) {
  int i = blockIdx.x * 256 + threadIdx.x;            // grid covers N*OUT exactly
  float v = hp[i];
  out[i] = v > 0.0f ? v : (__expf(v) - 1.0f);
}

// ---------------------------------------------------------------------------
// launch
// ---------------------------------------------------------------------------
extern "C" void kernel_launch(void* const* d_in, const int* in_sizes, int n_in,
                              void* d_out, int out_size, void* d_ws, size_t ws_size,
                              hipStream_t stream) {
  (void)in_sizes; (void)n_in; (void)out_size; (void)ws_size;
  const float* x  = (const float*)d_in[0];   // [N, 128]
  const float* W  = (const float*)d_in[1];   // [128, 64]
  const float* a  = (const float*)d_in[2];   // [128, 1]
  const int*   ei = (const int*)d_in[3];     // [2, E]
  const int* src = ei;
  const int* dst = ei + N_EDGES;
  float* out = (float*)d_out;                // [N, 64]

  // workspace layout (floats)
  float* ws   = (float*)d_ws;
  float* h    = ws;                                  // N*64
  float* hp   = h    + (size_t)N_NODES * OUT_DIM;    // N*64
  float* asrc = hp   + (size_t)N_NODES * OUT_DIM;    // N
  float* adst = asrc + N_NODES;                      // N
  float* e    = adst + N_NODES;                      // E (scores, then exp weights)
  unsigned* maxbits = (unsigned*)(e + N_EDGES);      // 1
  float* sum = (float*)(maxbits + 1);                // 1

  gat_init_kernel<<<(N_NODES * OUT_DIM) / 256, 256, 0, stream>>>(hp, maxbits, sum);
  gat_gemm_kernel<<<N_NODES / 16, 128, 0, stream>>>(x, W, h);
  gat_node_score_kernel<<<N_NODES / 8, 256, 0, stream>>>(h, a, asrc, adst);
  gat_edge_max_kernel<<<N_EDGES / 256, 256, 0, stream>>>(src, dst, asrc, adst, e, maxbits);
  gat_edge_exp_kernel<<<N_EDGES / 256, 256, 0, stream>>>(e, maxbits, sum);
  gat_scatter_kernel<<<N_EDGES / 8, 256, 0, stream>>>(src, dst, e, sum, h, hp);
  gat_elu_kernel<<<(N_NODES * OUT_DIM) / 256, 256, 0, stream>>>(hp, out);
}
